// SuperAttention_2757369004158
// MI455X (gfx1250) — matrix operand order repaired and checked by reading, not compiled, not run on hardware
//
#include <hip/hip_runtime.h>
#include <hip/hip_bf16.h>
#include <math.h>

// ---------------- CDNA5 WMMA plumbing (wave32, 16x16x32 bf16) ----------------
typedef __attribute__((ext_vector_type(16))) __bf16 bfx16;
typedef __attribute__((ext_vector_type(8)))  float  fx8;

#define BT       4
#define CDIM     128
#define NCELL    512      // 8*8*8 super tokens per batch
#define PDIM     64       // voxels per super token (4^3)
#define KPAD     32       // 27 neighbors padded to 32
#define K27      27
#define NHEADS   8
#define HD       16
#define EPSF     1e-12f
#define AFF_SCALE 0.08838834764831845f   // C^-0.5
#define Q_SCALE   0.25f                  // hd^-0.5

// ---- async global->LDS staging (gfx1250 GLOBAL_LOAD_ASYNC_TO_LDS_B128) ----
#if defined(__AMDGCN__) && __has_builtin(__builtin_amdgcn_global_load_async_to_lds_b128)
#define USE_ASYNC_LDS 1
#else
#define USE_ASYNC_LDS 0
#endif

typedef __attribute__((__vector_size__(16))) int i32x4;

__device__ __forceinline__ void lds_async_b128(void* sdst, const void* gsrc) {
#if USE_ASYNC_LDS
  __builtin_amdgcn_global_load_async_to_lds_b128(
      (__attribute__((address_space(1))) i32x4*)gsrc,
      (__attribute__((address_space(3))) i32x4*)sdst, 0, 0);
#else
  ((uint4*)sdst)[0] = ((const uint4*)gsrc)[0];
#endif
}

__device__ __forceinline__ void lds_async_wait() {
#if USE_ASYNC_LDS
  asm volatile("s_wait_asynccnt 0" ::: "memory");
#endif
}

__device__ __forceinline__ fx8 wmma_bf16(bfx16 a, bfx16 b, fx8 c) {
  return __builtin_amdgcn_wmma_f32_16x16x32_bf16(false, a, false, b, (short)0, c,
                                                 false, false);
}

// A fragment: rows striped over lanes (M = lane&15), K split per ISA layout.
__device__ __forceinline__ bfx16 frag_a(const __bf16* s, int ld, int row0, int k0) {
  int lane = threadIdx.x & 31;
  int hi   = lane >> 4;
  int r    = row0 + (lane & 15);
  bfx16 a;
#pragma unroll
  for (int e = 0; e < 16; ++e) {
    int k = k0 + hi * 8 + (e < 8 ? e : e + 8);
    a[e]  = s[r * ld + k];
  }
  return a;
}

// B fragment from [N][K] tile: N = lane&15, K split per ISA layout (same as A).
__device__ __forceinline__ bfx16 frag_b(const __bf16* s, int ld, int n0, int k0) {
  int lane = threadIdx.x & 31;
  int hi   = lane >> 4;
  int n    = n0 + (lane & 15);
  bfx16 b;
#pragma unroll
  for (int e = 0; e < 16; ++e) {
    int k = k0 + hi * 8 + (e < 8 ? e : e + 8);
    b[e]  = s[n * ld + k];
  }
  return b;
}

// ---------------- kernels ----------------

__global__ void k_zero(float* p, int n) {
  int i = blockIdx.x * 256 + threadIdx.x;
  if (i < n) p[i] = 0.f;
}

// super-token means (st0) + pixel tiles in bf16 [blk][p][c]
__global__ void k_block_stats(const float* __restrict__ x,
                              float* __restrict__ st0,
                              __bf16* __restrict__ pix) {
  int blk = blockIdx.x;           // bt*512 + m
  int bt = blk >> 9, m = blk & 511;
  int i = m >> 6, j = (m >> 3) & 7, l = m & 7;
  int c = threadIdx.x;            // 128 threads
  const float* xb = x + ((size_t)(bt * CDIM + c)) * 32768;
  __bf16* pw = pix + (size_t)blk * PDIM * CDIM + c;
  float sum = 0.f;
  for (int p = 0; p < PDIM; ++p) {
    int si = p >> 4, sj = (p >> 2) & 3, sl = p & 3;
    float v = xb[(i * 4 + si) * 1024 + (j * 4 + sj) * 32 + (l * 4 + sl)];
    sum += v;
    pw[(size_t)p * CDIM] = (__bf16)v;
  }
  st0[(size_t)blk * CDIM + c] = sum * (1.f / 64.f);
}

// aff = softmax_27(pixel @ su^T * scale); also writes both bf16 layouts + asum fold
__global__ void k_aff(const float* __restrict__ st_in,
                      const __bf16* __restrict__ pix,
                      __bf16* __restrict__ aff_pk,   // [blk][64p][32k]
                      __bf16* __restrict__ aff_kp,   // [blk][32k][64p]
                      float* __restrict__ asum) {
  __shared__ __align__(16) __bf16 sA[PDIM * CDIM];   // pixel [p][c]  16KB
  __shared__ __align__(16) __bf16 sB[KPAD * CDIM];   // su [kk][c]     8KB
  __shared__ float sL[PDIM * KPAD];                  // logits/probs   8KB
  int blk = blockIdx.x, bt = blk >> 9, m = blk & 511;
  int i = m >> 6, j = (m >> 3) & 7, l = m & 7;
  int tid = threadIdx.x;

  // async stage pixel tile (16KB, bit-exact) while the VALU gathers su below
  {
    const char* gp = (const char*)(pix + (size_t)blk * PDIM * CDIM);
    char* sp = (char*)sA;
    for (int it = tid; it < PDIM * CDIM * 2 / 16; it += 256)
      lds_async_b128(sp + it * 16, gp + it * 16);
  }

  for (int idx = tid; idx < KPAD * CDIM; idx += 256) {
    int kk = idx >> 7, c = idx & 127;
    float v = 0.f;
    if (kk < K27) {
      int di = kk / 9 - 1, dj = (kk / 3) % 3 - 1, dl = kk % 3 - 1;
      int ni = i + di, nj = j + dj, nl = l + dl;
      if ((unsigned)ni < 8u && (unsigned)nj < 8u && (unsigned)nl < 8u)
        v = st_in[((size_t)bt * NCELL + ni * 64 + nj * 8 + nl) * CDIM + c];
    }
    sB[kk * CDIM + c] = (__bf16)v;
  }
  lds_async_wait();
  __syncthreads();

  int wave = tid >> 5, lane = tid & 31, hi = lane >> 4;
  int rt = wave >> 1, nt = wave & 1;
  fx8 acc = {};
#pragma unroll
  for (int ks = 0; ks < 4; ++ks)
    acc = wmma_bf16(frag_a(sA, CDIM, rt * 16, ks * 32),
                    frag_b(sB, CDIM, nt * 16, ks * 32), acc);
#pragma unroll
  for (int g = 0; g < 8; ++g)
    sL[(rt * 16 + g + hi * 8) * KPAD + nt * 16 + (lane & 15)] = acc[g] * AFF_SCALE;
  __syncthreads();

  if (tid < PDIM) {
    float mx = -1e30f;
    for (int k = 0; k < K27; ++k) mx = fmaxf(mx, sL[tid * KPAD + k]);
    float sm = 0.f;
    for (int k = 0; k < K27; ++k) {
      float e = __expf(sL[tid * KPAD + k] - mx);
      sL[tid * KPAD + k] = e; sm += e;
    }
    float inv = 1.f / sm;
    __bf16* apk = aff_pk + (size_t)blk * PDIM * KPAD + tid * KPAD;
    __bf16* akp = aff_kp + (size_t)blk * KPAD * PDIM;
    for (int k = 0; k < KPAD; ++k) {
      float pv = (k < K27) ? sL[tid * KPAD + k] * inv : 0.f;
      sL[tid * KPAD + k] = pv;
      apk[k] = (__bf16)pv;
      akp[k * PDIM + tid] = (__bf16)pv;
    }
  }
  __syncthreads();

  if (tid < K27) {
    float cs = 0.f;
    for (int p = 0; p < PDIM; ++p) cs += sL[p * KPAD + tid];
    int di = tid / 9 - 1, dj = (tid / 3) % 3 - 1, dl = tid % 3 - 1;
    int ni = i + di, nj = j + dj, nl = l + dl;
    if ((unsigned)ni < 8u && (unsigned)nj < 8u && (unsigned)nl < 8u)
      atomicAdd(&asum[bt * NCELL + ni * 64 + nj * 8 + nl], cs);
  }
}

// stn = pixel^T @ aff, fold3 scatter-add into st_out [bt][cell][c]
__global__ void k_stn(const __bf16* __restrict__ pix,
                      const __bf16* __restrict__ aff_kp,
                      float* __restrict__ st_out) {
  __shared__ __align__(16) __bf16 sAT[CDIM * PDIM];  // [c][p] 16KB
  __shared__ __align__(16) __bf16 sB[KPAD * PDIM];   // [kk][p] 4KB
  int blk = blockIdx.x, bt = blk >> 9, m = blk & 511;
  int i = m >> 6, j = (m >> 3) & 7, l = m & 7;
  int tid = threadIdx.x;

  // async stage aff tile (bit-exact) while VALU transposes the pixel tile
  {
    const char* ga = (const char*)(aff_kp + (size_t)blk * KPAD * PDIM);
    char* sp = (char*)sB;
    for (int it = tid; it < KPAD * PDIM * 2 / 16; it += 256)
      lds_async_b128(sp + it * 16, ga + it * 16);
  }
  const __bf16* gp = pix + (size_t)blk * PDIM * CDIM;
  for (int idx = tid; idx < PDIM * CDIM; idx += 256) {
    int p = idx >> 7, c = idx & 127;
    sAT[c * PDIM + p] = gp[idx];
  }
  lds_async_wait();
  __syncthreads();

  int wave = tid >> 5, lane = tid & 31, hi = lane >> 4;
  for (int t = wave * 2; t < wave * 2 + 2; ++t) {
    int rt = t >> 1, nt = t & 1;
    fx8 acc = {};
#pragma unroll
    for (int ks = 0; ks < 2; ++ks)
      acc = wmma_bf16(frag_a(sAT, PDIM, rt * 16, ks * 32),
                      frag_b(sB, PDIM, nt * 16, ks * 32), acc);
    int kk = nt * 16 + (lane & 15);
    if (kk < K27) {
      int di = kk / 9 - 1, dj = (kk / 3) % 3 - 1, dl = kk % 3 - 1;
      int ni = i + di, nj = j + dj, nl = l + dl;
      if ((unsigned)ni < 8u && (unsigned)nj < 8u && (unsigned)nl < 8u) {
        float* dst = st_out + ((size_t)bt * NCELL + ni * 64 + nj * 8 + nl) * CDIM;
#pragma unroll
        for (int g = 0; g < 8; ++g) atomicAdd(&dst[rt * 16 + g + hi * 8], acc[g]);
      }
    }
  }
}

__global__ void k_norm(float* __restrict__ st, const float* __restrict__ asum) {
  int idx = blockIdx.x * 256 + threadIdx.x;
  if (idx >= BT * NCELL * CDIM) return;
  st[idx] /= (asum[idx / CDIM] + EPSF);
}

// qkv[n][o] = sum_c st[n][c] * w_qkv[o][c]  (q scaled by hd^-0.5), bf16 out
__global__ void k_qkv(const float* __restrict__ st, const float* __restrict__ w_qkv,
                      __bf16* __restrict__ qkv) {
  __shared__ __align__(16) __bf16 sA[64 * CDIM];   // 16KB
  __shared__ __align__(16) __bf16 sB[96 * CDIM];   // 24KB
  int bt = blockIdx.z, mbase = blockIdx.y * 64, obase = blockIdx.x * 96;
  int tid = threadIdx.x;
  for (int idx = tid; idx < 64 * CDIM; idx += 256) {
    int r = idx >> 7, c = idx & 127;
    sA[idx] = (__bf16)st[((size_t)bt * NCELL + mbase + r) * CDIM + c];
  }
  for (int idx = tid; idx < 96 * CDIM; idx += 256) {
    int r = idx >> 7, c = idx & 127;
    sB[idx] = (__bf16)w_qkv[(size_t)(obase + r) * CDIM + c];
  }
  __syncthreads();
  int wave = tid >> 5, lane = tid & 31, hi = lane >> 4;
  for (int t = wave * 3; t < wave * 3 + 3; ++t) {
    int rt = t / 6, nt = t % 6;
    fx8 acc = {};
#pragma unroll
    for (int ks = 0; ks < 4; ++ks)
      acc = wmma_bf16(frag_a(sA, CDIM, rt * 16, ks * 32),
                      frag_b(sB, CDIM, nt * 16, ks * 32), acc);
    int o = obase + nt * 16 + (lane & 15);
    float sc = (o < CDIM) ? Q_SCALE : 1.f;
#pragma unroll
    for (int g = 0; g < 8; ++g) {
      int r = mbase + rt * 16 + g + hi * 8;
      qkv[((size_t)bt * NCELL + r) * 384 + o] = (__bf16)(acc[g] * sc);
    }
  }
}

// per (bt,head,16 query rows): logits over all 512 keys + row softmax -> bf16 probs
__global__ void k_attn(const __bf16* __restrict__ qkv, __bf16* __restrict__ attnP) {
  __shared__ __align__(16) __bf16 sQ[16 * HD];
  __shared__ __align__(16) __bf16 sK[512 * HD];   // 16KB, rows are 32B
  __shared__ float sP[16 * 512];                  // 32KB
  __shared__ float sRA[16 * 16], sRB[16 * 16];
  int bt = blockIdx.z, head = blockIdx.y, mbase = blockIdx.x * 16;
  int tid = threadIdx.x;
  // async stage all keys: per row n, 32 contiguous bytes = two b128 chunks
  {
    const char* gk = (const char*)qkv;
    char* sp = (char*)sK;
    for (int it = tid; it < 512 * 2; it += 256) {
      int n = it >> 1, ch = it & 1;
      const char* g = gk + ((size_t)bt * NCELL + n) * 384 * 2 +
                      (CDIM + head * HD) * 2 + ch * 16;
      lds_async_b128(sp + it * 16, g);
    }
  }
  if (tid < 256) {
    int mm = tid >> 4, c = tid & 15;
    sQ[tid] = qkv[((size_t)bt * NCELL + mbase + mm) * 384 + head * HD + c];
  }
  lds_async_wait();
  __syncthreads();
  int wave = tid >> 5, lane = tid & 31, hi = lane >> 4;
  int nbase = wave * 64;
  for (int nt = 0; nt < 4; ++nt) {
    bfx16 a, b;
#pragma unroll
    for (int e = 0; e < 16; ++e) {
      int ka = hi * 8 + (e < 8 ? e : e + 8);
      a[e] = (ka < HD) ? sQ[(lane & 15) * HD + ka] : (__bf16)0.f;
      int kb = hi * 8 + (e < 8 ? e : e + 8);
      b[e] = (kb < HD) ? sK[(nbase + nt * 16 + (lane & 15)) * HD + kb] : (__bf16)0.f;
    }
    fx8 acc = {};
    acc = wmma_bf16(a, b, acc);
#pragma unroll
    for (int g = 0; g < 8; ++g)
      sP[(g + hi * 8) * 512 + nbase + nt * 16 + (lane & 15)] = acc[g];
  }
  __syncthreads();
  int r = tid >> 4, s = tid & 15;
  float mx = -1e30f;
  for (int n = s * 32; n < s * 32 + 32; ++n) mx = fmaxf(mx, sP[r * 512 + n]);
  sRA[r * 16 + s] = mx;
  __syncthreads();
  float rmax = -1e30f;
  for (int q = 0; q < 16; ++q) rmax = fmaxf(rmax, sRA[r * 16 + q]);
  float ps = 0.f;
  for (int n = s * 32; n < s * 32 + 32; ++n) {
    float e = __expf(sP[r * 512 + n] - rmax);
    sP[r * 512 + n] = e; ps += e;
  }
  sRB[r * 16 + s] = ps;
  __syncthreads();
  float rsum = 0.f;
  for (int q = 0; q < 16; ++q) rsum += sRB[r * 16 + q];
  float inv = 1.f / rsum;
  __bf16* gP = attnP + ((size_t)((bt * NHEADS + head) * 512 + mbase + r)) * 512;
  for (int n = s * 32; n < s * 32 + 32; ++n) gP[n] = (__bf16)(sP[r * 512 + n] * inv);
}

// O = P @ V^T per (bt,head); output heads concatenated in o_all[n][c]
__global__ void k_ov(const __bf16* __restrict__ attnP, const __bf16* __restrict__ qkv,
                     __bf16* __restrict__ o_all) {
  __shared__ __align__(16) __bf16 sA[128 * 128];  // 32KB probs chunk
  __shared__ __align__(16) __bf16 sV[HD * 128];   // 4KB  V^T chunk [c][n]
  int bt = blockIdx.z, head = blockIdx.y, mbase = blockIdx.x * 128;
  int tid = threadIdx.x;
  int wave = tid >> 5, lane = tid & 31, hi = lane >> 4;
  const char* gA = (const char*)(attnP +
                   ((size_t)((bt * NHEADS + head) * 512 + mbase)) * 512);
  fx8 acc = {};
  for (int kc = 0; kc < 4; ++kc) {
    // async stage probs chunk: row r (256B in LDS) from global row stride 1024B
    {
      char* sp = (char*)sA;
      for (int it = tid; it < 128 * 16; it += 256) {
        int r = it >> 4, ch = it & 15;
        lds_async_b128(sp + it * 16, gA + (size_t)r * 1024 + kc * 256 + ch * 16);
      }
    }
    for (int idx = tid; idx < HD * 128; idx += 256) {
      int c = idx >> 7, n = idx & 127;
      sV[idx] = qkv[((size_t)bt * NCELL + kc * 128 + n) * 384 + 2 * CDIM + head * HD + c];
    }
    lds_async_wait();
    __syncthreads();
#pragma unroll
    for (int ks = 0; ks < 4; ++ks)
      acc = wmma_bf16(frag_a(sA, 128, wave * 16, ks * 32),
                      frag_b(sV, 128, 0, ks * 32), acc);
    __syncthreads();
  }
  int c = lane & 15;
#pragma unroll
  for (int g = 0; g < 8; ++g) {
    int r = mbase + wave * 16 + g + hi * 8;
    o_all[((size_t)bt * NCELL + r) * CDIM + head * HD + c] = (__bf16)acc[g];
  }
}

// st_att[n][o] = o_all[n] @ w_proj^T + b_proj
__global__ void k_proj(const __bf16* __restrict__ o_all, const float* __restrict__ w_proj,
                       const float* __restrict__ b_proj, float* __restrict__ st_att) {
  __shared__ __align__(16) __bf16 sA[64 * CDIM];    // 16KB
  __shared__ __align__(16) __bf16 sB[CDIM * CDIM];  // 32KB
  int bt = blockIdx.y, mbase = blockIdx.x * 64;
  int tid = threadIdx.x;
  // async stage activations (bit-exact) while VALU converts w_proj to bf16
  {
    const char* gA = (const char*)(o_all + ((size_t)bt * NCELL + mbase) * CDIM);
    char* sp = (char*)sA;
    for (int it = tid; it < 64 * CDIM * 2 / 16; it += 256)
      lds_async_b128(sp + it * 16, gA + it * 16);
  }
  for (int idx = tid; idx < CDIM * CDIM; idx += 256) sB[idx] = (__bf16)w_proj[idx];
  lds_async_wait();
  __syncthreads();
  int wave = tid >> 5, lane = tid & 31, hi = lane >> 4;
  for (int t = wave * 4; t < wave * 4 + 4; ++t) {
    int rt = t >> 3, nt = t & 7;
    fx8 acc = {};
#pragma unroll
    for (int ks = 0; ks < 4; ++ks)
      acc = wmma_bf16(frag_a(sA, CDIM, rt * 16, ks * 32),
                      frag_b(sB, CDIM, nt * 16, ks * 32), acc);
    int o = nt * 16 + (lane & 15);
    float bias = b_proj[o];
#pragma unroll
    for (int g = 0; g < 8; ++g) {
      int r = mbase + rt * 16 + g + hi * 8;
      st_att[((size_t)bt * NCELL + r) * CDIM + o] = acc[g] + bias;
    }
  }
}

// pix = unfold(st_att) @ aff^T, scatter into output volume
__global__ void k_pix(const float* __restrict__ st_att, const __bf16* __restrict__ aff_pk,
                      float* __restrict__ out) {
  __shared__ __align__(16) __bf16 sSU[CDIM * KPAD]; // [c][kk] 8KB
  __shared__ __align__(16) __bf16 sB[PDIM * KPAD];  // [p][kk] 4KB
  int blk = blockIdx.x, bt = blk >> 9, m = blk & 511;
  int i = m >> 6, j = (m >> 3) & 7, l = m & 7;
  int tid = threadIdx.x;
  // async stage aff tile while VALU gathers/converts the su neighborhood
  {
    const char* ga = (const char*)(aff_pk + (size_t)blk * PDIM * KPAD);
    char* sp = (char*)sB;
    for (int it = tid; it < PDIM * KPAD * 2 / 16; it += 256)
      lds_async_b128(sp + it * 16, ga + it * 16);
  }
  for (int idx = tid; idx < CDIM * KPAD; idx += 256) {
    int c = idx >> 5, kk = idx & 31;
    float v = 0.f;
    if (kk < K27) {
      int di = kk / 9 - 1, dj = (kk / 3) % 3 - 1, dl = kk % 3 - 1;
      int ni = i + di, nj = j + dj, nl = l + dl;
      if ((unsigned)ni < 8u && (unsigned)nj < 8u && (unsigned)nl < 8u)
        v = st_att[((size_t)bt * NCELL + ni * 64 + nj * 8 + nl) * CDIM + c];
    }
    sSU[idx] = (__bf16)v;
  }
  lds_async_wait();
  __syncthreads();
  int wave = tid >> 5, lane = tid & 31, hi = lane >> 4;
  for (int t = wave * 4; t < wave * 4 + 4; ++t) {
    int rt = t >> 2, nt = t & 3;
    fx8 acc = {};
    acc = wmma_bf16(frag_a(sSU, KPAD, rt * 16, 0), frag_b(sB, KPAD, nt * 16, 0), acc);
    int p = nt * 16 + (lane & 15);
    int si = p >> 4, sj = (p >> 2) & 3, sl = p & 3;
    size_t base = (size_t)bt * CDIM * 32768 +
                  (size_t)(i * 4 + si) * 1024 + (j * 4 + sj) * 32 + (l * 4 + sl);
#pragma unroll
    for (int g = 0; g < 8; ++g) {
      int c = rt * 16 + g + hi * 8;
      out[base + (size_t)c * 32768] = acc[g];
    }
  }
}

// ---------------- launch ----------------
extern "C" void kernel_launch(void* const* d_in, const int* in_sizes, int n_in,
                              void* d_out, int out_size, void* d_ws, size_t ws_size,
                              hipStream_t stream) {
  const float* x      = (const float*)d_in[0];
  const float* w_qkv  = (const float*)d_in[1];
  const float* w_proj = (const float*)d_in[2];
  const float* b_proj = (const float*)d_in[3];
  float* out = (float*)d_out;

  char* ws = (char*)d_ws;
  size_t off = 0;
  auto take = [&](size_t bytes) -> void* {
    void* p = ws + off;
    off += (bytes + 255) & ~(size_t)255;
    return p;
  };
  __bf16* pix    = (__bf16*)take((size_t)BT * NCELL * PDIM * CDIM * 2); // 33.5MB
  float*  st0    = (float*) take((size_t)BT * NCELL * CDIM * 4);
  float*  st1    = (float*) take((size_t)BT * NCELL * CDIM * 4);
  float*  stf    = (float*) take((size_t)BT * NCELL * CDIM * 4);
  float*  st_att = (float*) take((size_t)BT * NCELL * CDIM * 4);
  float*  asum   = (float*) take((size_t)BT * NCELL * 4);
  __bf16* aff_pk = (__bf16*)take((size_t)BT * NCELL * PDIM * KPAD * 2); // 8.4MB
  __bf16* aff_kp = (__bf16*)take((size_t)BT * NCELL * KPAD * PDIM * 2); // 8.4MB
  __bf16* qkv    = (__bf16*)take((size_t)BT * NCELL * 384 * 2);
  __bf16* attnP  = (__bf16*)take((size_t)BT * NHEADS * 512 * 512 * 2);  // 16.8MB
  __bf16* o_all  = (__bf16*)take((size_t)BT * NCELL * CDIM * 2);

  const int nST = BT * NCELL * CDIM;  // 262144
  const int nAS = BT * NCELL;         // 2048

  k_block_stats<<<BT * NCELL, 128, 0, stream>>>(x, st0, pix);

  // ---- clustering iteration 0 ----
  k_zero<<<(nAS + 255) / 256, 256, 0, stream>>>(asum, nAS);
  k_aff<<<BT * NCELL, 256, 0, stream>>>(st0, pix, aff_pk, aff_kp, asum);
  k_zero<<<(nST + 255) / 256, 256, 0, stream>>>(st1, nST);
  k_stn<<<BT * NCELL, 256, 0, stream>>>(pix, aff_kp, st1);
  k_norm<<<(nST + 255) / 256, 256, 0, stream>>>(st1, asum);

  // ---- clustering iteration 1 (final aff/asum) ----
  k_zero<<<(nAS + 255) / 256, 256, 0, stream>>>(asum, nAS);
  k_aff<<<BT * NCELL, 256, 0, stream>>>(st1, pix, aff_pk, aff_kp, asum);

  // ---- final super-token aggregation ----
  k_zero<<<(nST + 255) / 256, 256, 0, stream>>>(stf, nST);
  k_stn<<<BT * NCELL, 256, 0, stream>>>(pix, aff_kp, stf);
  k_norm<<<(nST + 255) / 256, 256, 0, stream>>>(stf, asum);

  // ---- attention refinement on 512 super tokens ----
  k_qkv<<<dim3(4, 8, BT), 256, 0, stream>>>(stf, w_qkv, qkv);
  k_attn<<<dim3(32, NHEADS, BT), 256, 0, stream>>>(qkv, attnP);
  k_ov<<<dim3(4, NHEADS, BT), 256, 0, stream>>>(attnP, qkv, o_all);
  k_proj<<<dim3(8, BT), 256, 0, stream>>>(o_all, w_proj, b_proj, st_att);

  // ---- broadcast back to voxels ----
  k_pix<<<BT * NCELL, 256, 0, stream>>>(st_att, aff_pk, out);
}
